// XPLGru_78597901517032
// MI455X (gfx1250) — matrix operand order repaired and checked by reading, not compiled, not run on hardware
//
#include <hip/hip_runtime.h>

// ---------------------------------------------------------------------------
// GRU (16 steps, B=4096, H=1024) for MI455X / gfx1250, wave32 + WMMA.
// f32 emulated with bf16x3 split on V_WMMA_F32_16X16X32_BF16.
// Gate algebra: x_t == h_t for t>=1, so r,z use combined (W_ih+W_hh) weights:
//   r = sig(h@(Wih_r+Whh_r)^T + bir+bhr), z likewise,
//   n = tanh(h@Wih_n^T + bin + r * (h@Whh_n^T + bhn)).
// Weight B-tiles are staged into LDS with gfx1250 async copies, double-buffered.
// ---------------------------------------------------------------------------

typedef __bf16 bf16;
typedef __attribute__((ext_vector_type(16))) __bf16 v16bf;
typedef __attribute__((ext_vector_type(8)))  __bf16 v8bf;
typedef __attribute__((ext_vector_type(8)))  float  v8f;
typedef int v4i __attribute__((vector_size(16)));   // matches builtin param type

#define HDIM 1024
#define BATCH 4096

#define AS1 __attribute__((address_space(1)))
#define AS3 __attribute__((address_space(3)))

#if defined(__has_builtin)
# if __has_builtin(__builtin_amdgcn_global_load_async_to_lds_b128)
#  define HAVE_ASYNC_COPY 1
# endif
# if __has_builtin(__builtin_amdgcn_s_wait_asynccnt)
#  define HAVE_WAIT_ASYNC 1
# endif
#endif

template <int N>
__device__ __forceinline__ void wait_async() {
#if defined(HAVE_ASYNC_COPY)
# if defined(HAVE_WAIT_ASYNC)
    __builtin_amdgcn_s_wait_asynccnt(N);
# else
    asm volatile("s_wait_asynccnt %0" :: "i"(N) : "memory");
# endif
#endif
}

// copy 16B global -> LDS (async on gfx1250, sync fallback otherwise)
__device__ __forceinline__ void stage16(const bf16* gsrc, void* ldst) {
#if defined(HAVE_ASYNC_COPY)
    __builtin_amdgcn_global_load_async_to_lds_b128((v4i*)gsrc, (AS3 v4i*)ldst, 0, 0);
#else
    *(uint4*)ldst = *(const uint4*)gsrc;
#endif
}

// ---- fragment loaders (contiguous k = 16h+s, same for A and B, wave32) --------------

union AFrag { v16bf v; v8bf h[2]; };

// A (16x32 bf16): lanes 0-15: M=lane, K [0..15];
//                 lanes16-31: M=lane-16, K [16..31].
__device__ __forceinline__ v16bf load_a_frag(const bf16* __restrict__ p,
                                             int row, int k0, int lane) {
    const int kb = (lane >> 4) << 4;                    // 0 or 16
    const bf16* base = p + (size_t)row * HDIM + k0 + kb;
    AFrag a;
    a.h[0] = *(const v8bf*)(base);
    a.h[1] = *(const v8bf*)(base + 8);
    return a.v;
}

// B (32x16 bf16): lane n = lane&15 holds 16 contiguous K values,
// K offset 0 (lanes 0-15) or 16 (lanes 16-31).  B[k][n] = W[n][k] (row-major W).
__device__ __forceinline__ v16bf load_b_frag(const bf16* __restrict__ w,
                                             int row, int k0, int lane) {
    const int kb = (lane >> 4) << 4;                    // 0 or 16
    return *(const v16bf*)(w + (size_t)row * HDIM + k0 + kb);
}

// B fragment from an LDS row of 64B (32 bf16 of one W row), 80B stride outside.
__device__ __forceinline__ v16bf lds_b_frag(const char* rowp, int lane) {
    const char* p = rowp + ((lane >> 4) << 5);          // +0 or +32 bytes
    AFrag a;
    a.h[0] = *(const v8bf*)(p);
    a.h[1] = *(const v8bf*)(p + 16);
    return a.v;
}

__device__ __forceinline__ v8f wmma3(v16bf ahi, v16bf alo, v16bf bhi, v16bf blo, v8f c) {
    c = __builtin_amdgcn_wmma_f32_16x16x32_bf16(false, ahi, false, bhi, (short)0, c, false, false);
    c = __builtin_amdgcn_wmma_f32_16x16x32_bf16(false, ahi, false, blo, (short)0, c, false, false);
    c = __builtin_amdgcn_wmma_f32_16x16x32_bf16(false, alo, false, bhi, (short)0, c, false, false);
    return c;
}

// ---- f32 -> (hi, lo) bf16 plane split -------------------------------------

__global__ void split_kernel(const float* __restrict__ src,
                             bf16* __restrict__ hi, bf16* __restrict__ lo,
                             int count) {
    int i = blockIdx.x * blockDim.x + threadIdx.x;
    if (i < count) {
        float x = src[i];
        bf16 h = (bf16)x;
        hi[i] = h;
        lo[i] = (bf16)(x - (float)h);
    }
}

__global__ void combine_split_kernel(const float* __restrict__ a,
                                     const float* __restrict__ b,
                                     bf16* __restrict__ hi, bf16* __restrict__ lo,
                                     int count) {
    int i = blockIdx.x * blockDim.x + threadIdx.x;
    if (i < count) {
        float x = a[i] + b[i];
        bf16 h = (bf16)x;
        hi[i] = h;
        lo[i] = (bf16)(x - (float)h);
    }
}

// ---- fused gate GEMM + GRU cell -------------------------------------------
// Block: 256 thr = 8 waves as 4(M) x 2(N); wave tile 32x32; block tile 128x64.
// 4 gate GEMMs: g0=r(combined), g1=z(combined), g2=i_n(Wih), g3=h_n(Whh).
// B tiles (8 planes x 64 rows x 32 bf16) double-buffered in LDS via async copy.
__global__ __launch_bounds__(256) void gru_gate_kernel(
    const bf16* __restrict__ hhi, const bf16* __restrict__ hlo,
    const bf16* __restrict__ wc_hi, const bf16* __restrict__ wc_lo,   // (2H,H) r,z
    const bf16* __restrict__ wni_hi, const bf16* __restrict__ wni_lo, // (H,H) Wih n
    const bf16* __restrict__ wnh_hi, const bf16* __restrict__ wnh_lo, // (H,H) Whh n
    const float* __restrict__ b_ih, const float* __restrict__ b_hh,
    bf16* __restrict__ nhhi, bf16* __restrict__ nhlo,
    bf16* __restrict__ rhi, bf16* __restrict__ rlo,
    int first)
{
    __shared__ __align__(16) char smem[2][8][64][80];   // [buf][g*2+p][n][bytes]

    const int tid  = threadIdx.x;
    const int lane = tid & 31;
    const int wave = tid >> 5;
    const int mw = wave >> 1, nw = wave & 1;
    const int nblk = blockIdx.x * 64;
    const int m0 = blockIdx.y * 128 + mw * 32;

    // staging decode: thread -> (tile row sn, 16B chunk sck); gp = unrolled c
    const int sn = tid >> 2;                            // 0..63
    const int sck = tid & 3;                            // 0..3
    const bf16* gbase[8];
    gbase[0] = wc_hi  + (size_t)(nblk + sn) * HDIM;
    gbase[1] = wc_lo  + (size_t)(nblk + sn) * HDIM;
    gbase[2] = wc_hi  + (size_t)(HDIM + nblk + sn) * HDIM;
    gbase[3] = wc_lo  + (size_t)(HDIM + nblk + sn) * HDIM;
    gbase[4] = wni_hi + (size_t)(nblk + sn) * HDIM;
    gbase[5] = wni_lo + (size_t)(nblk + sn) * HDIM;
    gbase[6] = wnh_hi + (size_t)(nblk + sn) * HDIM;
    gbase[7] = wnh_lo + (size_t)(nblk + sn) * HDIM;

    auto stage = [&](int buf, int k0) {
        #pragma unroll
        for (int c = 0; c < 8; ++c)
            stage16(gbase[c] + k0 + sck * 8, &smem[buf][c][sn][sck * 16]);
    };

    v8f acc[2][4][2];                                   // [mf][gate][nf]
    #pragma unroll
    for (int f = 0; f < 2; ++f)
        #pragma unroll
        for (int g = 0; g < 4; ++g)
            #pragma unroll
            for (int nf = 0; nf < 2; ++nf)
                acc[f][g][nf] = (v8f){0.f,0.f,0.f,0.f,0.f,0.f,0.f,0.f};

    const int NK = HDIM / 32;
    stage(0, 0);
    for (int kc = 0; kc < NK; ++kc) {
        const int k0 = kc * 32;
        const int buf = kc & 1;
        if (kc + 1 < NK) {
            __syncthreads();           // all waves done reading buf^1 (iter kc-1)
            stage(buf ^ 1, k0 + 32);
            wait_async<8>();           // this wave's buf-cur copies complete
        } else {
            wait_async<0>();
        }
        __syncthreads();               // all waves' buf-cur copies complete

        v16bf ahi[2], alo[2];
        #pragma unroll
        for (int f = 0; f < 2; ++f) {
            const int row = m0 + f * 16 + (lane & 15);
            ahi[f] = load_a_frag(hhi, row, k0, lane);
            alo[f] = load_a_frag(hlo, row, k0, lane);
        }

        #pragma unroll
        for (int g = 0; g < 4; ++g) {
            if (first && g == 2) continue;     // step 0: x == 0 -> i_n = b_ih_n
            #pragma unroll
            for (int nf = 0; nf < 2; ++nf) {
                const int nl = nw * 32 + nf * 16 + (lane & 15);
                v16bf bhi = lds_b_frag(&smem[buf][2 * g + 0][nl][0], lane);
                v16bf blo = lds_b_frag(&smem[buf][2 * g + 1][nl][0], lane);
                #pragma unroll
                for (int f = 0; f < 2; ++f)
                    acc[f][g][nf] = wmma3(ahi[f], alo[f], bhi, blo, acc[f][g][nf]);
            }
        }
    }

    // Epilogue: GRU cell, write h_new and relu(h_new) split planes.
    #pragma unroll
    for (int nf = 0; nf < 2; ++nf) {
        const int n = nblk + nw * 32 + nf * 16 + (lane & 15);
        const float br   = b_ih[n] + b_hh[n];
        const float bz   = b_ih[HDIM + n] + b_hh[HDIM + n];
        const float bin_ = b_ih[2 * HDIM + n];
        const float bhn  = b_hh[2 * HDIM + n];
        #pragma unroll
        for (int f = 0; f < 2; ++f) {
            #pragma unroll
            for (int r = 0; r < 8; ++r) {
                const int m = m0 + f * 16 + r + ((lane >> 4) << 3);
                const size_t idx = (size_t)m * HDIM + n;
                float rg = 1.f / (1.f + __expf(-(acc[f][0][nf][r] + br)));
                float z  = 1.f / (1.f + __expf(-(acc[f][1][nf][r] + bz)));
                float ng = tanhf(acc[f][2][nf][r] + bin_ + rg * (acc[f][3][nf][r] + bhn));
                float hold = (float)hhi[idx] + (float)hlo[idx];
                float hnew = (1.f - z) * ng + z * hold;
                bf16 hh = (bf16)hnew;
                nhhi[idx] = hh;
                nhlo[idx] = (bf16)(hnew - (float)hh);
                float rr = hnew > 0.f ? hnew : 0.f;
                bf16 rh = (bf16)rr;
                rhi[idx] = rh;
                rlo[idx] = (bf16)(rr - (float)rh);
            }
        }
    }
}

// ---- output projection: out = relu(h_new) @ W_out^T + b_out ---------------
// Block: 8 waves stacked along M (tile 256x32); wave tile 32x32.
__global__ __launch_bounds__(256) void gru_out_kernel(
    const bf16* __restrict__ rhi, const bf16* __restrict__ rlo,
    const bf16* __restrict__ wo_hi, const bf16* __restrict__ wo_lo,
    const float* __restrict__ b_out, float* __restrict__ out)
{
    const int lane = threadIdx.x & 31;
    const int wave = threadIdx.x >> 5;
    const int m0 = blockIdx.y * 256 + wave * 32;
    const int n0 = blockIdx.x * 32;

    v8f acc[2][2];                                      // [mf][nf]
    #pragma unroll
    for (int f = 0; f < 2; ++f)
        #pragma unroll
        for (int nf = 0; nf < 2; ++nf)
            acc[f][nf] = (v8f){0.f,0.f,0.f,0.f,0.f,0.f,0.f,0.f};

    for (int k0 = 0; k0 < HDIM; k0 += 32) {
        v16bf bhi[2], blo[2];
        #pragma unroll
        for (int nf = 0; nf < 2; ++nf) {
            const int row = n0 + nf * 16 + (lane & 15);
            bhi[nf] = load_b_frag(wo_hi, row, k0, lane);
            blo[nf] = load_b_frag(wo_lo, row, k0, lane);
        }
        #pragma unroll
        for (int f = 0; f < 2; ++f) {
            const int row = m0 + f * 16 + (lane & 15);
            v16bf ahi = load_a_frag(rhi, row, k0, lane);
            v16bf alo = load_a_frag(rlo, row, k0, lane);
            #pragma unroll
            for (int nf = 0; nf < 2; ++nf)
                acc[f][nf] = wmma3(ahi, alo, bhi[nf], blo[nf], acc[f][nf]);
        }
    }

    #pragma unroll
    for (int nf = 0; nf < 2; ++nf) {
        const int n = n0 + nf * 16 + (lane & 15);
        const float bo = b_out[n];
        #pragma unroll
        for (int f = 0; f < 2; ++f)
            #pragma unroll
            for (int r = 0; r < 8; ++r) {
                const int m = m0 + f * 16 + r + ((lane >> 4) << 3);
                out[(size_t)m * HDIM + n] = acc[f][nf][r] + bo;
            }
    }
}

// ---------------------------------------------------------------------------

extern "C" void kernel_launch(void* const* d_in, const int* in_sizes, int n_in,
                              void* d_out, int out_size, void* d_ws, size_t ws_size,
                              hipStream_t stream) {
    (void)in_sizes; (void)n_in; (void)ws_size;
    const float* hidden = (const float*)d_in[0];  // (1, B, H)
    const float* W_ih   = (const float*)d_in[1];  // (3H, H)
    const float* W_hh   = (const float*)d_in[2];  // (3H, H)
    const float* b_ih   = (const float*)d_in[3];
    const float* b_hh   = (const float*)d_in[4];
    const float* W_out  = (const float*)d_in[5];  // (H, H)
    const float* b_out  = (const float*)d_in[6];
    float* out = (float*)d_out;

    const int SZ2 = 2 * HDIM * HDIM;     // 2,097,152 (r,z rows)
    const int SZ1 = HDIM * HDIM;         // 1,048,576
    const int SZH = BATCH * HDIM;        // 4,194,304

    char* ws = (char*)d_ws;
    size_t off = 0;
    auto alloc = [&](size_t elems) { bf16* p = (bf16*)(ws + off); off += elems * sizeof(bf16); return p; };
    bf16* wc_hi  = alloc(SZ2);  bf16* wc_lo  = alloc(SZ2);   // Wih+Whh (r,z)
    bf16* whr_hi = alloc(SZ2);  bf16* whr_lo = alloc(SZ2);   // Whh (r,z) for t=0
    bf16* wni_hi = alloc(SZ1);  bf16* wni_lo = alloc(SZ1);   // Wih n-gate
    bf16* wnh_hi = alloc(SZ1);  bf16* wnh_lo = alloc(SZ1);   // Whh n-gate
    bf16* wo_hi  = alloc(SZ1);  bf16* wo_lo  = alloc(SZ1);   // W_out
    bf16* hhi[2] = { alloc(SZH), alloc(SZH) };
    bf16* hlo[2] = { alloc(SZH), alloc(SZH) };
    bf16* rhi    = alloc(SZH);  bf16* rlo    = alloc(SZH);   // relu(h_new)

    int T = out_size / (BATCH * HDIM);
    if (T <= 0) T = 16;

    combine_split_kernel<<<(SZ2 + 255) / 256, 256, 0, stream>>>(W_ih, W_hh, wc_hi, wc_lo, SZ2);
    split_kernel<<<(SZ2 + 255) / 256, 256, 0, stream>>>(W_hh, whr_hi, whr_lo, SZ2);
    split_kernel<<<(SZ1 + 255) / 256, 256, 0, stream>>>(W_ih + (size_t)SZ2, wni_hi, wni_lo, SZ1);
    split_kernel<<<(SZ1 + 255) / 256, 256, 0, stream>>>(W_hh + (size_t)SZ2, wnh_hi, wnh_lo, SZ1);
    split_kernel<<<(SZ1 + 255) / 256, 256, 0, stream>>>(W_out, wo_hi, wo_lo, SZ1);
    split_kernel<<<(SZH + 255) / 256, 256, 0, stream>>>(hidden, hhi[0], hlo[0], SZH);

    dim3 blk(256);
    dim3 ggrd(HDIM / 64, BATCH / 128);   // (16, 32)
    dim3 ogrd(HDIM / 32, BATCH / 256);   // (32, 16)

    for (int t = 0; t < T; ++t) {
        const int cur = t & 1, nxt = cur ^ 1;
        gru_gate_kernel<<<ggrd, blk, 0, stream>>>(
            hhi[cur], hlo[cur],
            t == 0 ? whr_hi : wc_hi, t == 0 ? whr_lo : wc_lo,
            wni_hi, wni_lo, wnh_hi, wnh_lo,
            b_ih, b_hh, hhi[nxt], hlo[nxt], rhi, rlo, t == 0 ? 1 : 0);
        gru_out_kernel<<<ogrd, blk, 0, stream>>>(
            rhi, rlo, wo_hi, wo_lo, b_out, out + (size_t)t * BATCH * HDIM);
    }
}
